// GRLSTM_69853348102745
// MI455X (gfx1250) — compile-verified, matrix operand order checked by reading, not run on hardware
//
#include <hip/hip_runtime.h>
#include <hip/hip_bf16.h>
#include <math.h>

typedef _Float16 f16;
typedef __attribute__((ext_vector_type(16))) _Float16 v16h;
typedef __attribute__((ext_vector_type(8)))  _Float16 v8h;
typedef __attribute__((ext_vector_type(8)))  float    v8f;

union V16U { v16h v; v8h h[2]; };

// Load a 16-half WMMA fragment as two contiguous 16B chunks:
// elements 0..7 at p[0..7], elements 8..15 at p[16..23]
// (caller passes p = row_base + k0 + hi*8 per the CDNA5 16-bit A/B layout).
__device__ __forceinline__ v16h load_frag(const f16* p) {
    V16U u;
    u.h[0] = *(const v8h*)(p);
    u.h[1] = *(const v8h*)(p + 16);
    return u.v;
}

__device__ __forceinline__ float sigmoidf(float x) { return 1.0f / (1.0f + expf(-x)); }

__device__ __forceinline__ void atomicMaxF(float* addr, float val) {
    int old = __float_as_int(*addr);
    while (__int_as_float(old) < val) {
        int prev = atomicCAS((int*)addr, old, __float_as_int(val));
        if (prev == old) break;
        old = prev;
    }
}

// ---------------------------------------------------------------------------
// Generic WMMA GEMM:  out[M,Nout] = act(A[M,K](f16) @ WT[Nout,K](f16)^T + bias)
// 8 waves / block, one 16-row strip per wave, A-fragments hoisted across nt.
// ---------------------------------------------------------------------------
template<int K, bool RELU, bool OUT_F16>
__global__ __launch_bounds__(256) void wmma_gemm(
    const f16* __restrict__ A, int lda,
    const f16* __restrict__ WT,
    const float* __restrict__ bias,
    float* __restrict__ outF, f16* __restrict__ outH, int ldo,
    int M, int Nout)
{
    const int lane  = threadIdx.x & 31;
    const int wave  = threadIdx.x >> 5;
    const int mtile = blockIdx.x * 8 + wave;
    if (mtile * 16 >= M) return;
    const int m0 = mtile * 16;
    const int lm = lane & 15;
    const int hi = lane >> 4;
    constexpr int KC = K / 32;

    v16h afr[KC];
    const f16* arow = A + (size_t)(m0 + lm) * lda;
#pragma unroll
    for (int kc = 0; kc < KC; ++kc)
        afr[kc] = load_frag(arow + kc * 32 + hi * 8);

    for (int nt = 0; nt < (Nout >> 4); ++nt) {
        const int col = nt * 16 + lm;
        const float bv = bias ? bias[col] : 0.0f;
        v8f acc;
#pragma unroll
        for (int i = 0; i < 8; ++i) acc[i] = bv;

        const f16* brow = WT + (size_t)col * K;
#pragma unroll
        for (int kc = 0; kc < KC; ++kc) {
            v16h bfr = load_frag(brow + kc * 32 + hi * 8);
            acc = __builtin_amdgcn_wmma_f32_16x16x32_f16(
                false, afr[kc], false, bfr, (short)0, acc, false, false);
        }
#pragma unroll
        for (int r = 0; r < 8; ++r) {
            const int row = m0 + r + hi * 8;         // C/D layout: M=r (+8 for hi lanes)
            float v = acc[r];
            if (RELU) v = v > 0.0f ? v : 0.0f;
            if (OUT_F16) outH[(size_t)row * ldo + col] = (f16)v;
            else         outF[(size_t)row * ldo + col] = v;
        }
    }
}

// ---------------------------------------------------------------------------
// Persistent single-workgroup LSTM layer (B=64, T=64, D=128).
// Uses the full 320KB WGP LDS:
//   [  0 ..  32KB) sA   : x_t || h_t  (f16 [64][256])
//   [ 32 ..  64KB) sC   : cell state  (f32 [64][128])
//   [ 64 .. 320KB) Wlds : Wih||Whh    (f16 [512][256]) - resident all 64 steps
// 32 waves: wave = (mtile 0..3, chanblock 0..7); each wave computes the
// i/f/g/o 16x16 gate tiles for its (rows, channel-block) so the cell update
// is register-local.  The LDS weight base pointer is laundered through an
// opaque asm per time step so LLVM cannot hoist the (loop-invariant)
// B-fragment loads out of the loop and spill them to scratch.
// ---------------------------------------------------------------------------
__global__ __launch_bounds__(1024) void lstm_layer(
    const float* __restrict__ X,      // [64,64,128] layer input
    const f16*   __restrict__ Wcat,   // [512,256] = [Wih | Whh] f16, row=[out], col=[K]
    const float* __restrict__ bih,    // [512]
    const float* __restrict__ bhh,    // [512]
    float* __restrict__ OUT)          // [64,64,128]
{
    extern __shared__ char smem[];
    f16   (*sA)[256]   = (f16 (*)[256])smem;                    // 32KB
    float (*sC)[128]   = (float (*)[128])(smem + 64 * 256 * 2); // 32KB
    f16*   Wlds        = (f16*)(smem + 65536);                  // 256KB

    const int tid = threadIdx.x;

    // preload layer weights into LDS (256KB, 16B per thread per iter)
    {
        const uint4* src = (const uint4*)Wcat;
        uint4* dst = (uint4*)Wlds;
        for (int i = tid; i < (512 * 256 * 2) / 16; i += 1024) dst[i] = src[i];
    }
    for (int i = tid; i < 64 * 128; i += 1024) {
        sC[i >> 7][i & 127] = 0.0f;
        sA[i >> 7][128 + (i & 127)] = (f16)0.0f;                // h0 = 0
    }

    const int lane = tid & 31, wave = tid >> 5;
    const int mt = wave >> 3, cb = wave & 7;
    const int m0 = mt * 16, lm = lane & 15, hi = lane >> 4;
    const int ch = cb * 16 + lm;

    float bsum[4];
#pragma unroll
    for (int g = 0; g < 4; ++g) bsum[g] = bih[g * 128 + ch] + bhh[g * 128 + ch];

#pragma unroll 1
    for (int t = 0; t < 64; ++t) {
        __syncthreads();          // preload/h-writes visible; prev reads done
        for (int i = tid; i < 64 * 128; i += 1024) {
            sA[i >> 7][i & 127] = (f16)X[((size_t)(i >> 7) * 64 + t) * 128 + (i & 127)];
            if (t < 63)           // prefetch next step's x rows into cache
                __builtin_prefetch(&X[((size_t)(i >> 7) * 64 + t + 1) * 128 + (i & 127)], 0, 2);
        }
        __syncthreads();

        // opaque zero: makes Wlds addresses loop-variant to the optimizer
        int wz = 0;
        asm volatile("" : "+v"(wz));
        const f16* wb = Wlds + wz;

        v8f acc[4];
#pragma unroll
        for (int g = 0; g < 4; ++g) {
#pragma unroll
            for (int i = 0; i < 8; ++i) acc[g][i] = bsum[g];
        }

#pragma unroll
        for (int kc = 0; kc < 8; ++kc) {                        // K = 256 (x || h)
            v16h af = load_frag(&sA[m0 + lm][kc * 32 + hi * 8]);
#pragma unroll
            for (int g = 0; g < 4; ++g) {
                const f16* brow = wb + (size_t)(g * 128 + ch) * 256;
                v16h bf = load_frag(brow + kc * 32 + hi * 8);
                acc[g] = __builtin_amdgcn_wmma_f32_16x16x32_f16(
                    false, af, false, bf, (short)0, acc[g], false, false);
            }
        }

        __syncthreads();                                        // all waves done reading sA
#pragma unroll
        for (int r = 0; r < 8; ++r) {
            const int b = m0 + r + hi * 8;
            const float iv = acc[0][r], fv = acc[1][r], gv = acc[2][r], ov = acc[3][r];
            const float cold = sC[b][ch];
            const float cn = sigmoidf(fv) * cold + sigmoidf(iv) * tanhf(gv);
            const float hn = sigmoidf(ov) * tanhf(cn);
            sC[b][ch] = cn;
            sA[b][128 + ch] = (f16)hn;
            OUT[((size_t)b * 64 + t) * 128 + ch] = hn;
        }
    }
}

// softmax over time per channel, mean over channels, y = out*(1+a);
// then either residual+relu into Xnext, or final timestep into d_out.
__global__ __launch_bounds__(128) void attn_kernel(
    const float* __restrict__ OUT, const float* __restrict__ bemb,
    float* __restrict__ Xnext, float* __restrict__ dout, int is_final)
{
    __shared__ float sa[64];
    const int b = blockIdx.x, ch = threadIdx.x;
    float vals[64];
    float mx = -1e30f;
#pragma unroll
    for (int t = 0; t < 64; ++t) {
        vals[t] = OUT[((size_t)b * 64 + t) * 128 + ch];
        mx = fmaxf(mx, vals[t]);
    }
    float den = 0.0f;
#pragma unroll
    for (int t = 0; t < 64; ++t) den += expf(vals[t] - mx);
    if (ch < 64) sa[ch] = 0.0f;
    __syncthreads();
    const float inv = 1.0f / (den * 128.0f);
#pragma unroll
    for (int t = 0; t < 64; ++t) atomicAdd(&sa[t], expf(vals[t] - mx) * inv);
    __syncthreads();
    if (is_final) {
        dout[(size_t)b * 128 + ch] = vals[63] * (1.0f + sa[63]);
    } else {
#pragma unroll
        for (int t = 0; t < 64; ++t) {
            const float y = vals[t] * (1.0f + sa[t]);
            Xnext[((size_t)b * 64 + t) * 128 + ch] =
                bemb[((size_t)b * 64 + t) * 128 + ch] + fmaxf(y, 0.0f);
        }
    }
}

// ------------------------- small helper kernels ----------------------------
__global__ void fill_f32(float* p, float v, long n) {
    long t = blockIdx.x * (long)blockDim.x + threadIdx.x;
    if (t < n) p[t] = v;
}
__global__ void cvt_f16(const float* __restrict__ s, f16* __restrict__ d, long n) {
    long t = blockIdx.x * (long)blockDim.x + threadIdx.x;
    if (t < n) d[t] = (f16)s[t];
}
__global__ void transpose_to_f16(const float* __restrict__ src, f16* __restrict__ dst,
                                 int K, int Ncols) {   // src[K,N] -> dst[N,K]
    long t = blockIdx.x * (long)blockDim.x + threadIdx.x;
    if (t >= (long)K * Ncols) return;
    int k = (int)(t / Ncols), n = (int)(t % Ncols);
    dst[(size_t)n * K + k] = (f16)src[t];
}
__global__ void build_wcat(const float* __restrict__ Wih, const float* __restrict__ Whh,
                           f16* __restrict__ dst, long total) {  // [3][512][256]
    long t = blockIdx.x * (long)blockDim.x + threadIdx.x;
    if (t >= total) return;
    long l = t / (512 * 256);
    long r = (t / 256) % 512;
    long c = t % 256;
    float v = (c < 128) ? Wih[(l * 512 + r) * 128 + c]
                        : Whh[(l * 512 + r) * 128 + (c - 128)];
    dst[t] = (f16)v;
}
__global__ void node_att(const float* __restrict__ xp,
                         const float* __restrict__ asrc, const float* __restrict__ adst,
                         float* __restrict__ as_, float* __restrict__ ad_, int N) {
    int t = blockIdx.x * blockDim.x + threadIdx.x;
    if (t >= N * 8) return;
    const int n = t >> 3, h = t & 7;
    const float* xr = xp + (size_t)n * 128 + h * 16;
    float s = 0.0f, d = 0.0f;
#pragma unroll
    for (int c = 0; c < 16; ++c) { float v = xr[c]; s += v * asrc[h * 16 + c]; d += v * adst[h * 16 + c]; }
    as_[t] = s; ad_[t] = d;
}
__global__ void edge_alpha(const int* __restrict__ src, const int* __restrict__ dst,
                           const float* __restrict__ as_, const float* __restrict__ ad_,
                           float* __restrict__ alpha, float* __restrict__ mb, long Etot) {
    long t = blockIdx.x * (long)blockDim.x + threadIdx.x;
    if (t >= Etot * 8) return;
    long e = t >> 3; int h = (int)(t & 7);
    int s = src[e], d = dst[e];
    float a = as_[(size_t)s * 8 + h] + ad_[(size_t)d * 8 + h];
    a = a > 0.0f ? a : 0.2f * a;                 // leaky_relu(0.2)
    alpha[t] = a;
    atomicMaxF(&mb[(size_t)d * 8 + h], a);
}
__global__ void edge_exp(const int* __restrict__ dst, float* __restrict__ alpha,
                         const float* __restrict__ mb, float* __restrict__ den, long Etot) {
    long t = blockIdx.x * (long)blockDim.x + threadIdx.x;
    if (t >= Etot * 8) return;
    long e = t >> 3; int h = (int)(t & 7);
    int d = dst[e];
    float ev = expf(alpha[t] - mb[(size_t)d * 8 + h]);
    alpha[t] = ev;
    atomicAdd(&den[(size_t)d * 8 + h], ev);
}
__global__ void edge_acc(const int* __restrict__ src, const int* __restrict__ dst,
                         const float* __restrict__ alpha, const float* __restrict__ den,
                         const float* __restrict__ xp, float* __restrict__ out, long Etot) {
    long tid = blockIdx.x * (long)blockDim.x + threadIdx.x;
    long e = tid >> 5;
    if (e >= Etot) return;
    const int lane = (int)(tid & 31);
    const int s = src[e], d = dst[e];
    const int c0 = lane * 4, h = c0 >> 4;
    const float coef = alpha[e * 8 + h] / fmaxf(den[(size_t)d * 8 + h], 1e-16f);
    const float4 xv = *(const float4*)(xp + (size_t)s * 128 + c0);
    float* o = out + (size_t)d * 128 + c0;
    atomicAdd(o + 0, coef * xv.x);
    atomicAdd(o + 1, coef * xv.y);
    atomicAdd(o + 2, coef * xv.z);
    atomicAdd(o + 3, coef * xv.w);
}
__global__ void build_ewcat1(const float* __restrict__ o1, const float* __restrict__ o2,
                             const float* __restrict__ b1, const float* __restrict__ b2,
                             f16* __restrict__ dst, long n128) {
    long t = blockIdx.x * (long)blockDim.x + threadIdx.x;
    if (t >= n128) return;
    long n = t >> 7; int c = (int)(t & 127);
    dst[(size_t)n * 256 + c]       = (f16)(o1[t] + b1[c]);
    dst[(size_t)n * 256 + 128 + c] = (f16)(o2[t] + b2[c]);
}
__global__ __launch_bounds__(128) void fea_kernel(const float* __restrict__ fea_x,
                                                  const float* __restrict__ feaW,
                                                  const float* __restrict__ feab,
                                                  f16* __restrict__ ewcat2, int N) {
    const int n = blockIdx.x, d = threadIdx.x;
    if (n >= N) return;
    const float* fr = fea_x + (size_t)n * 8;
    float ss = 0.0f;
#pragma unroll
    for (int k = 0; k < 8; ++k) ss += fr[k] * fr[k];
    const float inv = 1.0f / fmaxf(sqrtf(ss), 1e-12f);
    float v = feab[d];
#pragma unroll
    for (int k = 0; k < 8; ++k) v += fr[k] * inv * feaW[k * 128 + d];
    ewcat2[(size_t)n * 256 + 128 + d] = (f16)fmaxf(v, 0.0f);
}
__global__ void gather_emb(const int* __restrict__ bx, const float* __restrict__ ewfin,
                           float* __restrict__ bemb, float* __restrict__ xcur, long n) {
    long t = blockIdx.x * (long)blockDim.x + threadIdx.x;
    if (t >= n) return;
    long bt = t >> 7; int c = (int)(t & 127);
    float v = ewfin[(size_t)bx[bt] * 128 + c];
    bemb[t] = v; xcur[t] = v;
}

// ---------------------------------------------------------------------------
extern "C" void kernel_launch(void* const* d_in, const int* in_sizes, int n_in,
                              void* d_out, int out_size, void* d_ws, size_t ws_size,
                              hipStream_t stream) {
    (void)n_in; (void)out_size; (void)ws_size;
    const float* poi      = (const float*)d_in[0];
    const float* fea_x    = (const float*)d_in[1];
    const int*   batch_x  = (const int*)  d_in[2];
    const int*   edge1    = (const int*)  d_in[3];
    const int*   edge2    = (const int*)  d_in[4];
    const float* gat1_W   = (const float*)d_in[5];
    const float* gat1_as  = (const float*)d_in[6];
    const float* gat1_ad  = (const float*)d_in[7];
    const float* gat1_b   = (const float*)d_in[8];
    const float* gat2_W   = (const float*)d_in[9];
    const float* gat2_as  = (const float*)d_in[10];
    const float* gat2_ad  = (const float*)d_in[11];
    const float* gat2_b   = (const float*)d_in[12];
    const float* fea_W    = (const float*)d_in[13];
    const float* fea_b    = (const float*)d_in[14];
    const float* m1W1     = (const float*)d_in[15];
    const float* m1b1     = (const float*)d_in[16];
    const float* m1W2     = (const float*)d_in[17];
    const float* m1b2     = (const float*)d_in[18];
    const float* m2W1     = (const float*)d_in[19];
    const float* m2b1     = (const float*)d_in[20];
    const float* m2W2     = (const float*)d_in[21];
    const float* m2b2     = (const float*)d_in[22];
    const float* lstm_Wih = (const float*)d_in[23];
    const float* lstm_Whh = (const float*)d_in[24];
    const float* lstm_bih = (const float*)d_in[25];
    const float* lstm_bhh = (const float*)d_in[26];

    const long N    = in_sizes[0] / 128;
    const long Etot = in_sizes[3] / 2;
    const long BT   = in_sizes[2];          // 64*64

    char* base = (char*)d_ws;
    size_t off = 0;
    auto alloc = [&](size_t bytes) -> void* {
        void* p = base + off; off += (bytes + 255) & ~(size_t)255; return p;
    };
    f16*   poiH   = (f16*)  alloc((size_t)N * 128 * 2);
    f16*   gWT1   = (f16*)  alloc(128 * 128 * 2);
    f16*   gWT2   = (f16*)  alloc(128 * 128 * 2);
    f16*   m1W1T  = (f16*)  alloc(256 * 128 * 2);
    f16*   m1W2T  = (f16*)  alloc(128 * 128 * 2);
    f16*   m2W1T  = (f16*)  alloc(256 * 128 * 2);
    f16*   m2W2T  = (f16*)  alloc(128 * 128 * 2);
    f16*   wcat   = (f16*)  alloc((size_t)3 * 512 * 256 * 2);
    float* xp     = (float*)alloc((size_t)N * 128 * 4);
    float* asb    = (float*)alloc((size_t)N * 8 * 4);
    float* adb    = (float*)alloc((size_t)N * 8 * 4);
    float* mb     = (float*)alloc((size_t)N * 8 * 4);
    float* den    = (float*)alloc((size_t)N * 8 * 4);
    float* alpha  = (float*)alloc((size_t)Etot * 8 * 4);
    float* out1   = (float*)alloc((size_t)N * 128 * 4);
    float* out2   = (float*)alloc((size_t)N * 128 * 4);
    f16*   ewcat1 = (f16*)  alloc((size_t)N * 256 * 2);
    f16*   hid    = (f16*)  alloc((size_t)N * 128 * 2);
    f16*   ewcat2 = (f16*)  alloc((size_t)N * 256 * 2);
    float* ewfin  = (float*)alloc((size_t)N * 128 * 4);
    float* bemb   = (float*)alloc((size_t)BT * 128 * 4);
    float* xcur   = (float*)alloc((size_t)BT * 128 * 4);
    float* lout   = (float*)alloc((size_t)BT * 128 * 4);

    auto cdiv = [](long a, long b) -> int { return (int)((a + b - 1) / b); };
    const int gemm_blocks = cdiv(N / 16, 8);

    // --- weight/activation prep (f16, transposed to [out,in]) ---
    cvt_f16<<<cdiv(N * 128, 256), 256, 0, stream>>>(poi, poiH, N * 128);
    transpose_to_f16<<<cdiv(128 * 128, 256), 256, 0, stream>>>(gat1_W, gWT1, 128, 128);
    transpose_to_f16<<<cdiv(128 * 128, 256), 256, 0, stream>>>(gat2_W, gWT2, 128, 128);
    transpose_to_f16<<<cdiv(256 * 128, 256), 256, 0, stream>>>(m1W1, m1W1T, 256, 128);
    transpose_to_f16<<<cdiv(128 * 128, 256), 256, 0, stream>>>(m1W2, m1W2T, 128, 128);
    transpose_to_f16<<<cdiv(256 * 128, 256), 256, 0, stream>>>(m2W1, m2W1T, 256, 128);
    transpose_to_f16<<<cdiv(128 * 128, 256), 256, 0, stream>>>(m2W2, m2W2T, 128, 128);
    build_wcat<<<cdiv(3L * 512 * 256, 256), 256, 0, stream>>>(lstm_Wih, lstm_Whh, wcat, 3L * 512 * 256);

    // --- GAT 1 and 2 (sequential; reuse xp/as/ad/m/den/alpha) ---
    for (int g = 0; g < 2; ++g) {
        const f16*   WT   = g == 0 ? gWT1 : gWT2;
        const float* avs  = g == 0 ? gat1_as : gat2_as;
        const float* avd  = g == 0 ? gat1_ad : gat2_ad;
        const int*   src  = g == 0 ? edge1 : edge2;
        const int*   dst  = g == 0 ? edge1 + Etot : edge2 + Etot;
        float*       outg = g == 0 ? out1 : out2;

        wmma_gemm<128, false, false><<<gemm_blocks, 256, 0, stream>>>(
            poiH, 128, WT, nullptr, xp, nullptr, 128, (int)N, 128);
        node_att<<<cdiv(N * 8, 256), 256, 0, stream>>>(xp, avs, avd, asb, adb, (int)N);
        fill_f32<<<cdiv(N * 8, 256), 256, 0, stream>>>(mb, -1e30f, N * 8);
        fill_f32<<<cdiv(N * 8, 256), 256, 0, stream>>>(den, 0.0f, N * 8);
        fill_f32<<<cdiv(N * 128, 256), 256, 0, stream>>>(outg, 0.0f, N * 128);
        edge_alpha<<<cdiv(Etot * 8, 256), 256, 0, stream>>>(src, dst, asb, adb, alpha, mb, Etot);
        edge_exp<<<cdiv(Etot * 8, 256), 256, 0, stream>>>(dst, alpha, mb, den, Etot);
        edge_acc<<<cdiv(Etot * 32, 256), 256, 0, stream>>>(src, dst, alpha, den, xp, outg, Etot);
    }

    // --- MLPs ---
    build_ewcat1<<<cdiv(N * 128, 256), 256, 0, stream>>>(out1, out2, gat1_b, gat2_b, ewcat1, N * 128);
    wmma_gemm<256, true, true><<<gemm_blocks, 256, 0, stream>>>(
        ewcat1, 256, m1W1T, m1b1, nullptr, hid, 128, (int)N, 128);
    wmma_gemm<128, false, true><<<gemm_blocks, 256, 0, stream>>>(
        hid, 128, m1W2T, m1b2, nullptr, ewcat2, 256, (int)N, 128);
    fea_kernel<<<(int)N, 128, 0, stream>>>(fea_x, fea_W, fea_b, ewcat2, (int)N);
    wmma_gemm<256, true, true><<<gemm_blocks, 256, 0, stream>>>(
        ewcat2, 256, m2W1T, m2b1, nullptr, hid, 128, (int)N, 128);
    wmma_gemm<128, false, false><<<gemm_blocks, 256, 0, stream>>>(
        hid, 128, m2W2T, m2b2, ewfin, nullptr, 128, (int)N, 128);

    // --- gather + 3x (LSTM + attention) ---
    gather_emb<<<cdiv(BT * 128, 256), 256, 0, stream>>>(batch_x, ewfin, bemb, xcur, BT * 128);
    const size_t lstm_smem = 64 * 256 * 2 + 64 * 128 * 4 + 512 * 256 * 2;  // 320KB WGP LDS
    for (int l = 0; l < 3; ++l) {
        lstm_layer<<<1, 1024, lstm_smem, stream>>>(
            xcur, wcat + (size_t)l * 512 * 256,
            lstm_bih + l * 512, lstm_bhh + l * 512, lout);
        attn_kernel<<<64, 128, 0, stream>>>(lout, bemb, xcur, (float*)d_out, l == 2 ? 1 : 0);
    }
}
